// ITreeLSTMCell_81235011437264
// MI455X (gfx1250) — matrix operand order repaired and checked by reading, not compiled, not run on hardware
//
#include <hip/hip_runtime.h>
#include <stdint.h>

// ---------------------------------------------------------------------------
// TreeLSTM cell: h,c,x_f = f(x @ [W_iou;W_f]^T, Uh_sum, fc_sum)
// bf16 WMMA (V_WMMA_F32_16X16X32_BF16) GEMM w/ fp32 accumulate, fused gates.
// Memory-bound target ~70us @ 23.3 TB/s; weights via TDM tensor_load_to_lds.
// ---------------------------------------------------------------------------

#define USE_TDM  1

#define N_ROWS   200000
#define H        256
#define XS       256            // K dimension
#define M_TILE   64             // rows per block
#define CT       32             // columns per segment per block (8 col groups)
#define WROWS    128            // 4 segments * CT local weight rows
#define WPAD_B   528            // 512B row + 16B pad -> 4-bank rotation in LDS
#define SW_BYTES (WROWS * WPAD_B)   // 67584
#define SX_BYTES (M_TILE * WPAD_B)  // 33792
#define SMEM_BYTES (SW_BYTES + SX_BYTES) // 101376 (<320KB WGP LDS, 3 blocks/WGP)

typedef __attribute__((ext_vector_type(16))) __bf16    v16bf;
typedef __attribute__((ext_vector_type(8)))  float     v8f;
typedef __attribute__((ext_vector_type(4)))  float     v4f;
typedef __attribute__((ext_vector_type(8)))  uint32_t  v8u;
typedef __attribute__((ext_vector_type(4)))  uint32_t  v4u;
typedef __attribute__((ext_vector_type(2)))  uint32_t  v2u;

union Frag { v4u q[2]; v16bf v; };

__device__ __forceinline__ uint32_t bf16pk(float a, float b) {
  // round-to-nearest-even f32 -> bf16, packed pair
  uint32_t ua = __float_as_uint(a);
  uint32_t ub = __float_as_uint(b);
  ua = (ua + 0x7FFFu + ((ua >> 16) & 1u)) >> 16;
  ub = (ub + 0x7FFFu + ((ub >> 16) & 1u)) & 0xFFFF0000u;
  return ua | ub;
}

// ---- kernel 1: convert [W_iou;W_f] f32 -> bf16 [1024][256] in workspace ----
__global__ __launch_bounds__(256) void wcvt_kernel(const float* __restrict__ Wiou,
                                                   const float* __restrict__ Wf,
                                                   uint32_t* __restrict__ wbf) {
  int idx = blockIdx.x * 256 + threadIdx.x;      // float4 index, 65536 total
  const int IOU4 = (3 * H * XS) / 4;             // 49152
  v4f v = (idx < IOU4) ? ((const v4f*)Wiou)[idx]
                       : ((const v4f*)Wf)[idx - IOU4];
  v2u p;
  p.x = bf16pk(v.x, v.y);
  p.y = bf16pk(v.z, v.w);
  ((v2u*)wbf)[idx] = p;
}

// ---- kernel 2: fused GEMM (bf16 WMMA) + TreeLSTM gate epilogue -------------
__global__ __launch_bounds__(256) void treelstm_kernel(
    const float*    __restrict__ x,      // [N,256]  (L2-reused 8x -> RT)
    const float*    __restrict__ Uh,     // [N,768]  (stream -> NT)
    const float*    __restrict__ fc,     // [N,256]  (stream -> NT)
    const float*    __restrict__ b_iou,  // [768]
    const float*    __restrict__ b_f,    // [256]
    const uint16_t* __restrict__ wbf,    // [1024][256] bf16 (L2-resident)
    float*          __restrict__ out)    // h | c | x_f, each [N,256] (NT)
{
  extern __shared__ __align__(16) unsigned char smem[];
  unsigned char* sW = smem;              // 128 x 528B padded bf16 weight rows
  unsigned char* sX = smem + SW_BYTES;   // 64  x 528B padded bf16 x rows

  const int tid  = threadIdx.x;
  const int g    = blockIdx.x;                // column group 0..7 (fastest)
  const int rb0  = blockIdx.y * M_TILE;       // row tile (slow dim)
  const int wave = tid >> 5;

#if USE_TDM
  // ---- weights via Tensor Data Mover: 3D tile (256 x 32rows x 4segments),
  //      16B LDS pad after every 512B row to match the padded layout.
  if (__builtin_amdgcn_readfirstlane(wave) == 0) {
    uint64_t ga  = (uint64_t)(uintptr_t)wbf + (uint64_t)g * (CT * XS * 2);
    uint32_t lds = (uint32_t)(uintptr_t)sW;   // LDS byte address (low 32 bits)
    v4u g0;
    g0.x = 1u;                                         // count=1, user mode
    g0.y = lds;                                        // lds_addr
    g0.z = (uint32_t)ga;                               // global_addr[31:0]
    g0.w = (uint32_t)((ga >> 32) & 0x1FFFFFFu)         // global_addr[56:32]
         | (2u << 30);                                 // type=2 (image)
    v8u g1;
    g1[0] = (1u << 16)       // data_size = 2 bytes
          | (1u << 20)       // pad_enable
          | (6u << 22)       // pad_interval: 2^(6+1)=128 dwords = 512B
          | (3u << 25);      // pad_amount: 4 dwords = 16B
    g1[1] = (uint32_t)XS << 16;          // tensor_dim0 = 256        [79:48]
    g1[2] = (uint32_t)H  << 16;          // tensor_dim1 = 256 rows   [111:80]
    g1[3] = (uint32_t)XS << 16;          // tile_dim0   = 256        [127:112]
    g1[4] = (uint32_t)CT | (4u << 16);   // tile_dim1=32, tile_dim2=4
    g1[5] = (uint32_t)XS;                // tensor_dim0_stride = 256 [207:160]
    g1[6] = 0u;                          // tensor_dim1_stride[15:0] = 0
    g1[7] = 1u;                          // tensor_dim1_stride = 65536 elems
    v4u g2 = {4u, 0u, 0u, 0u};           // tensor_dim2 = 4 segments
    v4u g3 = {0u, 0u, 0u, 0u};
    asm volatile("tensor_load_to_lds %0, %1, %2, %3"
                 :: "s"(g0), "s"(g1), "s"(g2), "s"(g3)
                 : "memory");
  }
#else
  // Manual weight fill: local row lr = s*32+rr <= global row s*256 + g*32 + rr
  #pragma unroll
  for (int i = 0; i < 16; ++i) {
    int idx  = tid + i * 256;                 // uint4 index, 4096 total
    int lr   = idx >> 5;
    int c    = idx & 31;
    int grow = ((lr >> 5) * H) + g * CT + (lr & 31);
    v4u w = ((const v4u*)(wbf + (size_t)grow * XS))[c];
    *(v4u*)(sW + lr * WPAD_B + c * 16) = w;
  }
#endif

  // ---- fill x tile LDS (f32 -> bf16), overlapped with the TDM transfer ----
  #pragma unroll
  for (int i = 0; i < 16; ++i) {
    int idx = tid + i * 256;                  // float4 index, 4096 total
    int r   = idx >> 6;
    int c4  = idx & 63;
    v4f v = ((const v4f*)(x + (size_t)(rb0 + r) * XS))[c4];
    v2u p; p.x = bf16pk(v.x, v.y); p.y = bf16pk(v.z, v.w);
    *(v2u*)(sX + r * WPAD_B + c4 * 8) = p;
  }

#if USE_TDM
  if (__builtin_amdgcn_readfirstlane(wave) == 0)
    __builtin_amdgcn_s_wait_tensorcnt(0);
#endif
  __syncthreads();

  const int lane = tid & 31;
  const int msub = wave >> 1;     // row sub-tile 0..3
  const int ch   = wave & 1;      // column half 0..1
  const int ln16 = lane & 15;
  const int lhi  = lane >> 4;

  // A (x) lane base: row = msub*16 + (lane&15); hi-lanes offset +16B (K+8)
  const unsigned char* aRow = sX + (msub * 16 + ln16) * WPAD_B + lhi * 16;
  // B (W) lane base: local row = s*32 + ch*16 + (lane&15); hi-lanes +16B (K+8)
  const unsigned char* bRow = sW + (ch * 16 + ln16) * WPAD_B + lhi * 16;

  v8f acc[4] = {};                // segments: i, o, u, f

  #pragma unroll
  for (int ks = 0; ks < 8; ++ks) {        // K = 256 in steps of 32
    Frag a;
    a.q[0] = *(const v4u*)(aRow + ks * 64);        // K base..+7 / +8..15
    a.q[1] = *(const v4u*)(aRow + ks * 64 + 32);   // K +16..23 / +24..31
    #pragma unroll
    for (int s = 0; s < 4; ++s) {
      Frag b;
      const unsigned char* bp = bRow + s * (CT * WPAD_B) + ks * 64;
      b.q[0] = *(const v4u*)(bp);
      b.q[1] = *(const v4u*)(bp + 32);
      acc[s] = __builtin_amdgcn_wmma_f32_16x16x32_bf16(
          /*neg_a=*/false, a.v, /*neg_b=*/false, b.v,
          /*c_mod=*/(short)0, acc[s], /*reuse_a=*/false, /*reuse_b=*/false);
    }
  }

  // ---- epilogue: fused gates; streaming traffic is non-temporal -----------
  const int col = g * CT + ch * 16 + ln16;         // 0..255
  const int rb  = rb0 + msub * 16 + lhi * 8;       // row for acc element j=0

  const float bi  = b_iou[col];
  const float bo  = b_iou[H + col];
  const float bu  = b_iou[2 * H + col];
  const float bfv = b_f[col];

  const float* uhp = Uh  + (size_t)rb * (3 * H) + col;
  const float* fcp = fc  + (size_t)rb * H + col;
  float*       hp  = out + (size_t)rb * H + col;
  float*       cp  = hp  + (size_t)N_ROWS * H;
  float*       xfp = cp  + (size_t)N_ROWS * H;

  #pragma unroll
  for (int j = 0; j < 8; ++j) {
    float xi = acc[0][j] + bi + __builtin_nontemporal_load(uhp);
    float xo = acc[1][j] + bo + __builtin_nontemporal_load(uhp + H);
    float xu = acc[2][j] + bu + __builtin_nontemporal_load(uhp + 2 * H);
    float xf = acc[3][j] + bfv;

    float si = 1.0f / (1.0f + __expf(-xi));                 // sigmoid(i)
    float so = 1.0f / (1.0f + __expf(-xo));                 // sigmoid(o)
    float tu = 1.0f - 2.0f / (__expf(2.0f * xu) + 1.0f);    // tanh(u)
    float cc = si * tu + __builtin_nontemporal_load(fcp);
    float tc = 1.0f - 2.0f / (__expf(2.0f * cc) + 1.0f);    // tanh(c)
    float hh = so * tc;

    __builtin_nontemporal_store(hh, hp);
    __builtin_nontemporal_store(cc, cp);
    __builtin_nontemporal_store(xf, xfp);
    uhp += 3 * H; fcp += H; hp += H; cp += H; xfp += H;
  }
}

// ---------------------------------------------------------------------------
extern "C" void kernel_launch(void* const* d_in, const int* in_sizes, int n_in,
                              void* d_out, int out_size, void* d_ws, size_t ws_size,
                              hipStream_t stream) {
  const float* x    = (const float*)d_in[0];   // [N,256]
  const float* Uh   = (const float*)d_in[1];   // [N,768]
  const float* fc   = (const float*)d_in[2];   // [N,256]
  const float* Wiou = (const float*)d_in[3];   // [768,256]
  const float* biou = (const float*)d_in[4];   // [1,768]
  const float* Wf   = (const float*)d_in[5];   // [256,256]
  const float* bf   = (const float*)d_in[6];   // [1,256]
  float* out = (float*)d_out;

  uint16_t* wbf = (uint16_t*)d_ws;             // needs 512KB of workspace
  (void)in_sizes; (void)n_in; (void)out_size; (void)ws_size;

  // 1) weights f32 -> bf16 (L2-resident, 512KB)
  wcvt_kernel<<<256, 256, 0, stream>>>(Wiou, Wf, (uint32_t*)wbf);

  // 2) fused WMMA GEMM + gate epilogue.
  //    Column group is the FAST grid dim so the 8 blocks sharing an x tile
  //    dispatch back-to-back -> x stays in L2 (one HBM pass over x).
  dim3 grid(8, N_ROWS / M_TILE);
  treelstm_kernel<<<grid, 256, SMEM_BYTES, stream>>>(x, Uh, fc, biou, bf, wbf, out);
}
